// LagsAttention_4733053960341
// MI455X (gfx1250) — matrix operand order repaired and checked by reading, not compiled, not run on hardware
//
#include <hip/hip_runtime.h>

// ---------------------------------------------------------------------------
// Problem constants (fixed by the reference): B=8, s=192, d=512, L=8
// ---------------------------------------------------------------------------
constexpr int Bn  = 8;
constexpr int Ss  = 192;
constexpr int Dd  = 512;
constexpr int Ll  = 8;
constexpr int SEQ = Ss * Ll;   // 1536
constexpr int Hh  = 2048;

typedef __bf16 bf16;
typedef __attribute__((ext_vector_type(8)))  bf16  v8bf;
typedef __attribute__((ext_vector_type(16))) bf16  v16bf;
typedef __attribute__((ext_vector_type(8)))  float v8f;

// Async global->LDS path (gfx1250 GLOBAL_LOAD_ASYNC_TO_LDS_B128, ASYNCcnt).
// Parameter types per hipcc diagnostic: ptr to vector_size(16) int in AS1 / AS3.
#if __has_builtin(__builtin_amdgcn_global_load_async_to_lds_b128)
#define HAVE_ASYNC_LDS 1
typedef int vsi4 __attribute__((vector_size(4 * sizeof(int))));
typedef __attribute__((address_space(1))) vsi4 as1_vsi4;
typedef __attribute__((address_space(3))) vsi4 as3_vsi4;
#else
#define HAVE_ASYNC_LDS 0
#endif

__device__ __forceinline__ void wait_async_lds()
{
#if HAVE_ASYNC_LDS
#if __has_builtin(__builtin_amdgcn_s_wait_asynccnt)
    __builtin_amdgcn_s_wait_asynccnt(0);
#else
    asm volatile("s_wait_asynccnt 0" ::: "memory");
#endif
#endif
}

// ---------------------------------------------------------------------------
// Kernel 1: pack lags (B,s,d,L) -> x (B,S,d) as f32 + bf16 + bf16-transposed
// ---------------------------------------------------------------------------
__global__ __launch_bounds__(256)
void pack_x_kernel(const float* __restrict__ lags,
                   float* __restrict__ xf, bf16* __restrict__ xb,
                   bf16* __restrict__ xtb)
{
    long long idx = (long long)blockIdx.x * 256 + threadIdx.x;   // over B*S*d
    int dd = (int)(idx % Dd);
    long long t = idx / Dd;
    int q = (int)(t % SEQ);
    int b = (int)(t / SEQ);
    int si = q / Ll, l = q % Ll;
    float v = lags[(((long long)b * Ss + si) * Dd + dd) * Ll + l];
    xf[idx] = v;
    xb[idx] = (bf16)v;
    xtb[((long long)b * Dd + dd) * SEQ + q] = (bf16)v;
}

// ---------------------------------------------------------------------------
// Kernel 2: pack W1 (d,h) -> W1t (h,d) bf16 and W2 (h,d) -> W2t (d,h) bf16
// ---------------------------------------------------------------------------
__global__ __launch_bounds__(256)
void pack_w_kernel(const float* __restrict__ W1, const float* __restrict__ W2,
                   bf16* __restrict__ W1t, bf16* __restrict__ W2t)
{
    long long idx = (long long)blockIdx.x * 256 + threadIdx.x;   // over d*h
    // W1 stored [d][h]: idx = dd*Hh + hh
    int hh = (int)(idx % Hh);
    int dd = (int)(idx / Hh);
    W1t[(long long)hh * Dd + dd] = (bf16)W1[idx];
    // W2 stored [h][d]: idx = hh2*Dd + dd2
    int dd2 = (int)(idx % Dd);
    int hh2 = (int)(idx / Dd);
    W2t[(long long)dd2 * Hh + hh2] = (bf16)W2[idx];
}

// ---------------------------------------------------------------------------
// WMMA GEMM:  C[M,N] (f32) = alpha * A[M,K](bf16,row-major) * Bt[N,K]^T
// Block tile 128x128, BK=32, 8 waves (2 along M x 4 along N),
// each wave: 64x32 = 4x2 tiles of v_wmma_f32_16x16x32_bf16.
// Double-buffered LDS staging; async global->LDS when available.
// ---------------------------------------------------------------------------
#define TBM 128
#define TBN 128
#define TBK 32

__global__ __launch_bounds__(256)
void gemm_bt_kernel(const bf16* __restrict__ A, const bf16* __restrict__ Bt,
                    float* __restrict__ C,
                    int M, int N, int K,
                    long long sA, long long sB, long long sC,
                    float alpha)
{
    __shared__ __align__(128) bf16 As[2][TBM][TBK];
    __shared__ __align__(128) bf16 Bs[2][TBN][TBK];

    const int tid  = threadIdx.x;
    const int lane = tid & 31;
    const int wave = tid >> 5;
    const int bm = blockIdx.y * TBM;
    const int bn = blockIdx.x * TBN;

    A  += (long long)blockIdx.z * sA;
    Bt += (long long)blockIdx.z * sB;
    C  += (long long)blockIdx.z * sC;

    const int wm = (wave & 1) * 64;   // wave row offset inside block tile
    const int wn = (wave >> 1) * 32;  // wave col offset inside block tile

    const v8f zero = {0.f, 0.f, 0.f, 0.f, 0.f, 0.f, 0.f, 0.f};
    v8f acc[4][2];
#pragma unroll
    for (int i = 0; i < 4; ++i)
#pragma unroll
        for (int j = 0; j < 2; ++j) acc[i][j] = zero;

    // cooperative tile loader: 16B per thread per pass, 2 passes per matrix
    const int lr = tid >> 2;          // 0..63
    const int lc = (tid & 3) * 8;     // 0,8,16,24
    auto load_tile = [&](int buf, int k0) {
#pragma unroll
        for (int p = 0; p < 2; ++p) {
            int row = p * 64 + lr;
            const bf16* ga = &A[(long long)(bm + row) * K + k0 + lc];
            const bf16* gb = &Bt[(long long)(bn + row) * K + k0 + lc];
            bf16* la = &As[buf][row][lc];
            bf16* lb = &Bs[buf][row][lc];
#if HAVE_ASYNC_LDS
            __builtin_amdgcn_global_load_async_to_lds_b128(
                (as1_vsi4*)ga, (as3_vsi4*)la, 0, 0);
            __builtin_amdgcn_global_load_async_to_lds_b128(
                (as1_vsi4*)gb, (as3_vsi4*)lb, 0, 0);
#else
            *(v8bf*)la = *(const v8bf*)ga;
            *(v8bf*)lb = *(const v8bf*)gb;
#endif
        }
    };

    // A fragment (16x32 bf16): lanes 0-15 row r hold K 0-15,
    // lanes 16-31 row r hold K 16-31 (matches B fragment order)
    const int ar = lane & 15;
    const int akb = (lane >> 4) * 16;
    auto load_afrag = [&](int buf, int mt) -> v16bf {
        const bf16* p = &As[buf][wm + mt * 16 + ar][0];
        v8bf lo = *(const v8bf*)(p + akb);
        v8bf hi = *(const v8bf*)(p + akb + 8);
        return __builtin_shufflevector(lo, hi,
            0,1,2,3,4,5,6,7,8,9,10,11,12,13,14,15);
    };

    // B fragment (32x16 bf16): lane holds a contiguous K=16 run of one column
    // (lanes 0-15: K 0-15, lanes 16-31: K 16-31) -> one 32B LDS read
    const int bc  = lane & 15;
    const int bkb = (lane >> 4) * 16;
    auto load_bfrag = [&](int buf, int nt) -> v16bf {
        return *(const v16bf*)&Bs[buf][wn + nt * 16 + bc][bkb];
    };

    load_tile(0, 0);
    wait_async_lds();
    __syncthreads();

    const int nk = K / TBK;
    for (int kk = 0; kk < nk; ++kk) {
        const int buf = kk & 1;
        if (kk + 1 < nk) load_tile(buf ^ 1, (kk + 1) * TBK);
        if (kk + 2 < nk) {  // global_prefetch_b8 for the tile after next
            __builtin_prefetch(&A[(long long)(bm + lr) * K + (kk + 2) * TBK], 0, 1);
            __builtin_prefetch(&Bt[(long long)(bn + lr) * K + (kk + 2) * TBK], 0, 1);
        }

        v16bf bf0 = load_bfrag(buf, 0);
        v16bf bf1 = load_bfrag(buf, 1);
#pragma unroll
        for (int i = 0; i < 4; ++i) {
            v16bf af = load_afrag(buf, i);
            acc[i][0] = __builtin_amdgcn_wmma_f32_16x16x32_bf16(
                false, af, false, bf0, (short)0, acc[i][0], false, false);
            acc[i][1] = __builtin_amdgcn_wmma_f32_16x16x32_bf16(
                false, af, false, bf1, (short)0, acc[i][1], false, false);
        }
        wait_async_lds();
        __syncthreads();
    }

    // Epilogue: C/D layout -> lane col = lane&15, VGPR v = row v (lanes<16)
    // or row 8+v (lanes>=16)  (ISA 7.12.2)
    const int col  = lane & 15;
    const int rsel = (lane >> 4) * 8;
#pragma unroll
    for (int i = 0; i < 4; ++i) {
#pragma unroll
        for (int j = 0; j < 2; ++j) {
            const int gr0 = bm + wm + i * 16 + rsel;
            const int gc  = bn + wn + j * 16 + col;
#pragma unroll
            for (int v = 0; v < 8; ++v)
                C[(long long)(gr0 + v) * N + gc] = acc[i][j][v] * alpha;
        }
    }
}

// ---------------------------------------------------------------------------
// Softmax over rows of length SEQ (one wave32 per row, 48 elems/lane)
// scores f32 -> P bf16
// ---------------------------------------------------------------------------
__global__ __launch_bounds__(256)
void softmax_kernel(const float* __restrict__ Sc, bf16* __restrict__ P)
{
    const int wave = threadIdx.x >> 5, lane = threadIdx.x & 31;
    const long long row = (long long)blockIdx.x * 8 + wave;
    const float* src = Sc + row * SEQ;
    bf16* dst = P + row * SEQ;

    float v[SEQ / 32];
    float m = -3.402823466e+38f;
#pragma unroll
    for (int t = 0; t < SEQ / 32; ++t) {
        v[t] = src[lane + t * 32];
        m = fmaxf(m, v[t]);
    }
#pragma unroll
    for (int off = 16; off > 0; off >>= 1) m = fmaxf(m, __shfl_xor(m, off, 32));

    float sum = 0.f;
#pragma unroll
    for (int t = 0; t < SEQ / 32; ++t) { v[t] = __expf(v[t] - m); sum += v[t]; }
#pragma unroll
    for (int off = 16; off > 0; off >>= 1) sum += __shfl_xor(sum, off, 32);

    const float inv = 1.f / sum;
#pragma unroll
    for (int t = 0; t < SEQ / 32; ++t) dst[lane + t * 32] = (bf16)(v[t] * inv);
}

// ---------------------------------------------------------------------------
// q = LayerNorm(x + y) -> f32 + bf16 (one wave per row of Dd=512)
// ---------------------------------------------------------------------------
__global__ __launch_bounds__(256)
void add_ln_kernel(const float* __restrict__ xa, const float* __restrict__ xb,
                   const float* __restrict__ gamma, const float* __restrict__ beta,
                   float* __restrict__ qf, bf16* __restrict__ qb)
{
    const int wave = threadIdx.x >> 5, lane = threadIdx.x & 31;
    const long long row = (long long)blockIdx.x * 8 + wave;
    const float* pa = xa + row * Dd;
    const float* pb = xb + row * Dd;

    float v[Dd / 32];
    float sum = 0.f;
#pragma unroll
    for (int t = 0; t < Dd / 32; ++t) {
        int c = lane + t * 32;
        v[t] = pa[c] + pb[c];
        sum += v[t];
    }
#pragma unroll
    for (int off = 16; off > 0; off >>= 1) sum += __shfl_xor(sum, off, 32);
    const float mu = sum * (1.f / Dd);

    float vs = 0.f;
#pragma unroll
    for (int t = 0; t < Dd / 32; ++t) { float d = v[t] - mu; vs += d * d; }
#pragma unroll
    for (int off = 16; off > 0; off >>= 1) vs += __shfl_xor(vs, off, 32);
    const float inv = rsqrtf(vs * (1.f / Dd) + 1e-5f);

#pragma unroll
    for (int t = 0; t < Dd / 32; ++t) {
        int c = lane + t * 32;
        float o = (v[t] - mu) * inv * gamma[c] + beta[c];
        qf[row * Dd + c] = o;
        qb[row * Dd + c] = (bf16)o;
    }
}

// ---------------------------------------------------------------------------
// y = GELU(x + b1)  (exact, erf) : f32 -> bf16, elementwise over B*S*h
// ---------------------------------------------------------------------------
__global__ __launch_bounds__(256)
void bias_gelu_kernel(const float* __restrict__ ff1, const float* __restrict__ b1,
                      bf16* __restrict__ out)
{
    long long idx = (long long)blockIdx.x * 256 + threadIdx.x;
    int col = (int)(idx % Hh);
    float x = ff1[idx] + b1[col];
    float g = 0.5f * x * (1.f + erff(x * 0.70710678118654752f));
    out[idx] = (bf16)g;
}

// ---------------------------------------------------------------------------
// out = transpose_back(LayerNorm(q + ff2 + b2))  (one wave per row)
// ---------------------------------------------------------------------------
__global__ __launch_bounds__(256)
void final_ln_kernel(const float* __restrict__ qf, const float* __restrict__ ff2,
                     const float* __restrict__ b2,
                     const float* __restrict__ gamma, const float* __restrict__ beta,
                     float* __restrict__ out)
{
    const int wave = threadIdx.x >> 5, lane = threadIdx.x & 31;
    const long long row = (long long)blockIdx.x * 8 + wave;
    const float* pq = qf + row * Dd;
    const float* pf = ff2 + row * Dd;

    float v[Dd / 32];
    float sum = 0.f;
#pragma unroll
    for (int t = 0; t < Dd / 32; ++t) {
        int c = lane + t * 32;
        v[t] = pq[c] + pf[c] + b2[c];
        sum += v[t];
    }
#pragma unroll
    for (int off = 16; off > 0; off >>= 1) sum += __shfl_xor(sum, off, 32);
    const float mu = sum * (1.f / Dd);

    float vs = 0.f;
#pragma unroll
    for (int t = 0; t < Dd / 32; ++t) { float d = v[t] - mu; vs += d * d; }
#pragma unroll
    for (int off = 16; off > 0; off >>= 1) vs += __shfl_xor(vs, off, 32);
    const float inv = rsqrtf(vs * (1.f / Dd) + 1e-5f);

    const int q = (int)(row % SEQ);
    const int b = (int)(row / SEQ);
    const int si = q / Ll, l = q % Ll;
#pragma unroll
    for (int t = 0; t < Dd / 32; ++t) {
        int c = lane + t * 32;
        float o = (v[t] - mu) * inv * gamma[c] + beta[c];
        out[(((long long)b * Ss + si) * Dd + c) * Ll + l] = o;
    }
}

// ---------------------------------------------------------------------------
// Host-side launcher
// ---------------------------------------------------------------------------
extern "C" void kernel_launch(void* const* d_in, const int* in_sizes, int n_in,
                              void* d_out, int out_size, void* d_ws, size_t ws_size,
                              hipStream_t stream)
{
    const float* lags   = (const float*)d_in[0];
    const float* gamma1 = (const float*)d_in[1];
    const float* beta1  = (const float*)d_in[2];
    const float* W1     = (const float*)d_in[3];
    const float* b1     = (const float*)d_in[4];
    const float* W2     = (const float*)d_in[5];
    const float* b2     = (const float*)d_in[6];
    const float* gamma3 = (const float*)d_in[7];
    const float* beta3  = (const float*)d_in[8];
    float* out = (float*)d_out;

    const size_t NX  = (size_t)Bn * SEQ * Dd;   // 6,291,456
    const size_t NSS = (size_t)Bn * SEQ * SEQ;  // 18,874,368
    const size_t NH  = (size_t)Bn * SEQ * Hh;   // 25,165,824
    const size_t NW  = (size_t)Dd * Hh;         // 1,048,576

    char* w = (char*)d_ws;
    size_t off = 0;
    auto alloc = [&](size_t bytes) -> char* {
        char* p = w + off;
        off = (off + bytes + 255) & ~(size_t)255;
        return p;
    };
    float* x_f32  = (float*)alloc(NX * 4);
    bf16*  x_bf   = (bf16*) alloc(NX * 2);
    bf16*  xT_bf  = (bf16*) alloc(NX * 2);
    float* scores = (float*)alloc(NSS * 4);
    bf16*  P_bf   = (bf16*) alloc(NSS * 2);
    float* attn   = (float*)alloc(NX * 4);
    float* q_f32  = (float*)alloc(NX * 4);
    bf16*  q_bf   = (bf16*) alloc(NX * 2);
    bf16*  W1t    = (bf16*) alloc(NW * 2);
    bf16*  W2t    = (bf16*) alloc(NW * 2);
    float* ff1    = (float*)alloc(NH * 4);
    bf16*  gel    = (bf16*) alloc(NH * 2);
    float* ff2    = (float*)alloc(NX * 4);

    // 1) pack inputs
    pack_x_kernel<<<dim3((unsigned)(NX / 256)), 256, 0, stream>>>(lags, x_f32, x_bf, xT_bf);
    pack_w_kernel<<<dim3((unsigned)(NW / 256)), 256, 0, stream>>>(W1, W2, W1t, W2t);

    // 2) scores = x x^T / sqrt(d)   [B, S, S]
    gemm_bt_kernel<<<dim3(SEQ / TBN, SEQ / TBM, Bn), 256, 0, stream>>>(
        x_bf, x_bf, scores, SEQ, SEQ, Dd,
        (long long)SEQ * Dd, (long long)SEQ * Dd, (long long)SEQ * SEQ,
        0.04419417382415922f /* 1/sqrt(512) */);

    // 3) softmax rows -> P (bf16)
    softmax_kernel<<<dim3(Bn * SEQ / 8), 256, 0, stream>>>(scores, P_bf);

    // 4) attn_out = P x   [B, S, d]   (Bt = x^T, pre-packed)
    gemm_bt_kernel<<<dim3(Dd / TBN, SEQ / TBM, Bn), 256, 0, stream>>>(
        P_bf, xT_bf, attn, SEQ, Dd, SEQ,
        (long long)SEQ * SEQ, (long long)Dd * SEQ, (long long)SEQ * Dd, 1.0f);

    // 5) q = LN(x + attn_out)
    add_ln_kernel<<<dim3(Bn * SEQ / 8), 256, 0, stream>>>(
        x_f32, attn, gamma1, beta1, q_f32, q_bf);

    // 6) ff1 = q W1   [B*S, h]
    gemm_bt_kernel<<<dim3(Hh / TBN, (Bn * SEQ) / TBM, 1), 256, 0, stream>>>(
        q_bf, W1t, ff1, Bn * SEQ, Hh, Dd, 0, 0, 0, 1.0f);

    // 7) gelu(ff1 + b1) -> bf16
    bias_gelu_kernel<<<dim3((unsigned)(NH / 256)), 256, 0, stream>>>(ff1, b1, gel);

    // 8) ff2 = gelu W2   [B*S, d]
    gemm_bt_kernel<<<dim3(Dd / TBN, (Bn * SEQ) / TBM, 1), 256, 0, stream>>>(
        gel, W2t, ff2, Bn * SEQ, Dd, Hh, 0, 0, 0, 1.0f);

    // 9) out = LN(q + ff2 + b2), transposed back to (B, s, d, L)
    final_ln_kernel<<<dim3(Bn * SEQ / 8), 256, 0, stream>>>(
        q_f32, ff2, b2, gamma3, beta3, out);

    (void)in_sizes; (void)n_in; (void)out_size; (void)ws_size;
}
